// multi_head_31001073942744
// MI455X (gfx1250) — compile-verified, matrix operand order checked by reading, not run on hardware
//
#include <hip/hip_runtime.h>

typedef __attribute__((ext_vector_type(16))) __bf16 v16bf;
typedef __attribute__((ext_vector_type(8)))  float  v8f;

#define B_     2
#define M_     128
#define N_     128
#define HEADS  8
#define DH     64
#define MW_    16
#define NW_    16
#define NWIN   (MW_*NW_)          // 256 windows per (b,h)
#define KPAD   128                // 116 real keys padded to 128
#define KREAL  116

union Frag {
    v16bf v;
    unsigned short u[16];
    uint4 q[2];
};

__device__ __forceinline__ unsigned short f2bf(float f) {
    union { float f; unsigned u; } x; x.f = f;
    unsigned r = x.u + 0x7FFFu + ((x.u >> 16) & 1u);   // round-nearest-even
    return (unsigned short)(r >> 16);
}
__device__ __forceinline__ float bf2f(unsigned short h) {
    union { unsigned u; float f; } x; x.u = ((unsigned)h) << 16; return x.f;
}

__device__ __forceinline__ v8f wmma_bf16(const Frag& a, const Frag& b, v8f c) {
    return __builtin_amdgcn_wmma_f32_16x16x32_bf16(false, a.v, false, b.v, (short)0, c, false, false);
}

// ---- fragment loaders (16-bit A: lane m=L%16, k runs {base, base+16}+((L>=16)?8:0)) ----
__device__ __forceinline__ void load_a_bf16(Frag& f, const unsigned short* Am, int row0, int ldk, int k0, int lane) {
    const unsigned short* p = Am + (size_t)(row0 + (lane & 15)) * ldk + k0 + ((lane >= 16) ? 8 : 0);
    f.q[0] = *(const uint4*)(p);
    f.q[1] = *(const uint4*)(p + 16);
}
// B element (k, n) from a [n][k] bf16 matrix (k contiguous) -> two b128 loads
__device__ __forceinline__ void load_b_nk(Frag& f, const unsigned short* Bm, int col0, int ldk, int k0, int lane) {
    const unsigned short* p = Bm + (size_t)(col0 + (lane & 15)) * ldk + k0 + ((lane >= 16) ? 8 : 0);
    f.q[0] = *(const uint4*)(p);
    f.q[1] = *(const uint4*)(p + 16);
}
// A rows from windowed-attention bf16 layout [(b*8+h)*16384 + win*64 + q][d]
__device__ __forceinline__ void load_a_att(Frag& f, const unsigned short* A, int row0, int k0, int lane) {
    int r = row0 + (lane & 15);
    int b = r >> 14, m = (r >> 7) & 127, n = r & 127;
    size_t winq = ((size_t)(m >> 3) * NW_ + (n >> 3)) * 64 + ((m & 7) * 8 + (n & 7));
    int kb = k0 + ((lane >= 16) ? 8 : 0);
#pragma unroll
    for (int h = 0; h < 2; ++h) {
        int k = kb + h * 16;
        int hd = k >> 6, d = k & 63;
        const unsigned short* p = A + (((size_t)b * HEADS + hd) * 16384 + winq) * 64 + d;
        f.q[h] = *(const uint4*)(p);
    }
}

// ================= one-time packs: fp32 -> bf16 =================
__global__ void pack_rm_k(const float* in, unsigned short* out, int total8) {
    int t = blockIdx.x * blockDim.x + threadIdx.x;
    if (t >= total8) return;
    const float* p = in + (size_t)t * 8;
    union { unsigned short u[8]; uint4 q; } r;
#pragma unroll
    for (int e = 0; e < 8; ++e) r.u[e] = f2bf(p[e]);
    *(uint4*)(out + (size_t)t * 8) = r.q;
}
// W[k][n] fp32 (512x512) -> Wt[n][k] bf16 (transpose pack)
__global__ void pack_wt_k(const float* W, unsigned short* Wt) {
    int t = blockIdx.x * blockDim.x + threadIdx.x;   // 512*64
    int n = t >> 6, k0 = (t & 63) * 8;
    union { unsigned short u[8]; uint4 q; } r;
#pragma unroll
    for (int e = 0; e < 8; ++e) r.u[e] = f2bf(W[(size_t)(k0 + e) * 512 + n]);
    *(uint4*)(Wt + (size_t)n * 512 + k0) = r.q;
}

// ================= GEMM: C[R x 512] = A[R x 512] * Wt^T =================
// A bf16 row-major (or windowed att layout), Wt bf16 [n][k].
// 256 threads = 8 waves; wave -> 32x64 tile (2x4 WMMA); WG -> 128 x 128
template <bool ATT_A, bool OUT_BF16>
__global__ void gemm512_k(const void* Aptr, const unsigned short* Wt, void* Cptr, int R) {
    int lane = threadIdx.x & 31, wid = threadIdx.x >> 5;
    int rowBlocks = R >> 7;
    int wgRow = blockIdx.x % rowBlocks;
    int wgCol = blockIdx.x / rowBlocks;
    int r0 = wgRow * 128 + (wid & 3) * 32;
    int c0 = wgCol * 128 + (wid >> 2) * 64;
    v8f z = {0.f,0.f,0.f,0.f,0.f,0.f,0.f,0.f};
    v8f acc[2][4];
#pragma unroll
    for (int i = 0; i < 2; ++i)
#pragma unroll
        for (int j = 0; j < 4; ++j) acc[i][j] = z;
#pragma unroll 2
    for (int t = 0; t < 16; ++t) {
        int k0 = t * 32;
        Frag a[2], bb[4];
#pragma unroll
        for (int s = 0; s < 2; ++s) {
            if constexpr (ATT_A) load_a_att(a[s], (const unsigned short*)Aptr, r0 + s * 16, k0, lane);
            else                 load_a_bf16(a[s], (const unsigned short*)Aptr, r0 + s * 16, 512, k0, lane);
        }
#pragma unroll
        for (int j = 0; j < 4; ++j) load_b_nk(bb[j], Wt, c0 + j * 16, 512, k0, lane);
#pragma unroll
        for (int i = 0; i < 2; ++i)
#pragma unroll
            for (int j = 0; j < 4; ++j) acc[i][j] = wmma_bf16(a[i], bb[j], acc[i][j]);
    }
#pragma unroll
    for (int i = 0; i < 2; ++i)
#pragma unroll
        for (int j = 0; j < 4; ++j) {
            int cc = c0 + j * 16 + (lane & 15);
#pragma unroll
            for (int r = 0; r < 8; ++r) {
                int rr = r0 + i * 16 + ((lane >= 16) ? 8 : 0) + r;
                if constexpr (OUT_BF16) ((unsigned short*)Cptr)[(size_t)rr * 512 + cc] = f2bf(acc[i][j][r]);
                else                    ((float*)Cptr)[(size_t)rr * 512 + cc] = acc[i][j][r];
            }
        }
}

// ================= RoPE + window relayout for Q (bf16 in, bf16 out) =================
__global__ void rope_q_k(const unsigned short* Qp, const float* freq, unsigned short* qW, int total) {
    int t = blockIdx.x * blockDim.x + threadIdx.x;
    if (t >= total) return;
    int dp = t & 31; int rest = t >> 5;
    int n = rest & 127; rest >>= 7;
    int m = rest & 127; rest >>= 7;
    int h = rest & 7;   int b = rest >> 3;
    int de = dp * 2, dof = de + 1;
    int f = (dp < 16) ? dp : dp - 16;
    size_t fo = (((size_t)b * M_ + m) * N_ + n) * 128;
    float se = freq[fo + de],  so = freq[fo + dof];
    float oe = freq[fo + 64 + de], oo = freq[fo + 64 + dof];
    size_t qi = (((size_t)b * M_ + m) * N_ + n) * 512 + h * 64;
    float ve = bf2f(Qp[qi + de])  * se + oe;
    float vo = bf2f(Qp[qi + dof]) * so + oo;
    float pos = (float)((dp < 16) ? m : n);
    float ang = pos * __powf(10000.0f, -(float)f * (1.0f / 16.0f));
    float c, s; __sincosf(ang, &c, &s);
    size_t o = ((((size_t)b * HEADS + h) * NWIN + (m >> 3) * NW_ + (n >> 3)) * 64 + ((m & 7) * 8 + (n & 7))) * 64;
    qW[o + de]  = f2bf(c * ve - s * vo);
    qW[o + dof] = f2bf(s * ve + c * vo);
}

// cy_run per level from input cy (S_POOL=2 each level, 2*pad carry: +8, +6)
__device__ __forceinline__ int center_l(int c, int lvl) {
    int c0 = c >> 1;            if (lvl == 0) return c0;
    int c1 = (c0 + 8) >> 1;     if (lvl == 1) return c1;
    return (c1 + 6) >> 1;
}

// ================= gather regions + freq + RoPE for K and V =================
// kW: [win][key][d] (QK^T B-operand);  vW: [win][d][key] transposed (PV B-operand)
__global__ void gather_kv_k(const unsigned short* Kp, const unsigned short* Vp, const float* pm, const float* freq,
                            const int* cy, const int* cx,
                            unsigned short* kW, unsigned short* vW, float* maskK,
                            int lvl, int ml, int nl, int kk, int pad, int slotOff, int total) {
    int t = blockIdx.x * blockDim.x + threadIdx.x;
    if (t >= total) return;
    int dp = t & 31; int rest = t >> 5;
    int dx = rest % kk; rest /= kk;
    int dy = rest % kk; rest /= kk;
    int j = rest & 15; rest >>= 4;
    int i = rest & 15; rest >>= 4;
    int h = rest & 7;  int b = rest >> 3;

    int cyl = center_l(cy[i], lvl), cxl = center_l(cx[j], lvl);
    int YK = min(cyl + dy, ml - 1), XK = min(cxl + dx, nl - 1);
    int yk = YK - pad, xk = XK - pad;
    bool inb = (yk >= 0) && (xk >= 0);
    float mval = inb ? pm[((size_t)b * ml + yk) * nl + xk] : 0.0f;
    int yv = min(cyl + pad + dy, ml - 1) - pad;
    int xv = min(cxl + pad + dx, nl - 1) - pad;
    yv = max(yv, 0); xv = max(xv, 0);

    int de = dp * 2, dof = de + 1;
    int f = (dp < 16) ? dp : dp - 16;
    int posm = min(max(yk * 2, 0), M_ - 1);
    int posn = min(max(xk * 2, 0), N_ - 1);
    size_t fo = (((size_t)b * M_ + posm) * N_ + posn) * 128;
    float se = freq[fo + de], so = freq[fo + dof];
    float oe = freq[fo + 64 + de], oo = freq[fo + 64 + dof];
    float ang = (float)((dp < 16) ? posm : posn) * __powf(10000.0f, -(float)f * (1.0f / 16.0f));
    float c, s; __sincosf(ang, &c, &s);

    float kre = 0.f, kro = 0.f;
    if (inb) {
        size_t kb = (((size_t)b * ml + yk) * nl + xk) * 512 + h * 64;
        kre = bf2f(Kp[kb + de]); kro = bf2f(Kp[kb + dof]);
    }
    float ke = kre * se + oe, ko = kro * so + oo;
    size_t vb = (((size_t)b * ml + yv) * nl + xv) * 512 + h * 64;
    float vee = bf2f(Vp[vb + de]) * se + oe, voo = bf2f(Vp[vb + dof]) * so + oo;

    int slot = slotOff + dy * kk + dx;
    size_t win = ((size_t)b * HEADS + h) * NWIN + i * NW_ + j;
    size_t ok = (win * KPAD + slot) * 64;
    kW[ok + de]  = f2bf(c * ke - s * ko);
    kW[ok + dof] = f2bf(s * ke + c * ko);
    vW[(win * 64 + de)  * KPAD + slot] = f2bf(c * vee - s * voo);
    vW[(win * 64 + dof) * KPAD + slot] = f2bf(s * vee + c * voo);
    if (h == 0 && dp == 0)
        maskK[((size_t)b * NWIN + i * NW_ + j) * KPAD + slot] = (mval != 0.0f) ? 1.0f : 0.0f;
}

// zero pad slots 116..127 (kW row-layout, vW transposed layout)
__global__ void pad_tail_k(unsigned short* kW, unsigned short* vW, float* maskK) {
    int t = blockIdx.x * blockDim.x + threadIdx.x;
    const int totalKV = B_ * HEADS * NWIN * (KPAD - KREAL) * 64;
    if (t < totalKV) {
        int d = t & 63; int rest = t >> 6;
        int sl = rest % (KPAD - KREAL); int win = rest / (KPAD - KREAL);
        kW[((size_t)win * KPAD + KREAL + sl) * 64 + d] = 0;
        vW[((size_t)win * 64 + d) * KPAD + KREAL + sl] = 0;
    }
    if (t < B_ * NWIN * (KPAD - KREAL)) {
        int sl = t % (KPAD - KREAL); int bw = t / (KPAD - KREAL);
        maskK[(size_t)bw * KPAD + KREAL + sl] = 0.0f;
    }
}

// ================= per-window attention: softmax(QK^T/8 + mask) V =================
// grid = B*H*NWIN blocks, 128 threads (4 waves); wave w -> 16 query rows
__global__ void attn_win_k(const unsigned short* qW, const unsigned short* kW, const unsigned short* vW,
                           const float* maskK, const unsigned char* mask_q, unsigned short* attW) {
    __shared__ float sS[4][16][KPAD];
    __shared__ unsigned short sP[4][16][KPAD];
    __shared__ float sSum[4][16];

    int win = blockIdx.x;
    int ij = win & (NWIN - 1);
    int b = win >> 11;                       // win = ((b*8+h)*256 + ij)
    int lane = threadIdx.x & 31, w = threadIdx.x >> 5;
    const unsigned short* Qw = qW + (size_t)win * 64 * 64;
    const unsigned short* Kw = kW + (size_t)win * KPAD * 64;
    const unsigned short* Vw = vW + (size_t)win * 64 * KPAD;
    const float* mk = maskK + ((size_t)b * NWIN + ij) * KPAD;
    const unsigned char* mq = mask_q + ((size_t)b * NWIN + ij) * 64;
    int q0 = w * 16;

    // pull the V tile toward the WGP while QK^T runs (global_prefetch_b8)
    __builtin_prefetch(Vw + (size_t)threadIdx.x * 128, 0, 1);

    Frag aQ[2];
    load_a_bf16(aQ[0], Qw, q0, 64, 0, lane);
    load_a_bf16(aQ[1], Qw, q0, 64, 32, lane);

    unsigned qbits = 0;
#pragma unroll
    for (int r = 0; r < 8; ++r)
        qbits |= (mq[q0 + ((lane >= 16) ? 8 : 0) + r] != 0) ? (1u << r) : 0u;

    v8f z = {0.f,0.f,0.f,0.f,0.f,0.f,0.f,0.f};
#pragma unroll
    for (int kt = 0; kt < 8; ++kt) {
        v8f acc = z;
        Frag bK;
        load_b_nk(bK, Kw, kt * 16, 64, 0, lane);  acc = wmma_bf16(aQ[0], bK, acc);
        load_b_nk(bK, Kw, kt * 16, 64, 32, lane); acc = wmma_bf16(aQ[1], bK, acc);
        int key = kt * 16 + (lane & 15);
        bool mv = mk[key] != 0.f;
#pragma unroll
        for (int r = 0; r < 8; ++r) {
            int mrow = ((lane >= 16) ? 8 : 0) + r;
            float sc = (mv && ((qbits >> r) & 1u)) ? acc[r] * 0.125f : -3.0e38f;
            sS[w][mrow][key] = sc;
        }
    }
    __syncthreads();
    {
        // all 32 lanes: lane&15 -> row, lane>>4 -> which 64-key half; float4 LDS reads
        int r = lane & 15, half = lane >> 4;
        const float4* rowp = (const float4*)&sS[w][r][half * 64];
        float mx = -3.4e38f;
#pragma unroll
        for (int c4 = 0; c4 < 16; ++c4) {
            float4 v4 = rowp[c4];
            mx = fmaxf(mx, fmaxf(fmaxf(v4.x, v4.y), fmaxf(v4.z, v4.w)));
        }
        mx = fmaxf(mx, __shfl_xor(mx, 16, 32));
        float sum = 0.f;
#pragma unroll
        for (int c4 = 0; c4 < 16; ++c4) {
            float4 v4 = rowp[c4];
            float e0 = __expf(v4.x - mx), e1 = __expf(v4.y - mx);
            float e2 = __expf(v4.z - mx), e3 = __expf(v4.w - mx);
            sum += (e0 + e1) + (e2 + e3);
            union { unsigned short us[4]; uint2 q; } pk;
            pk.us[0] = f2bf(e0); pk.us[1] = f2bf(e1); pk.us[2] = f2bf(e2); pk.us[3] = f2bf(e3);
            *(uint2*)&sP[w][r][half * 64 + c4 * 4] = pk.q;
        }
        sum += __shfl_xor(sum, 16, 32);
        if (half == 0) sSum[w][r] = sum;
    }
    __syncthreads();

    v8f acc[4] = { z, z, z, z };
#pragma unroll
    for (int t = 0; t < 4; ++t) {
        Frag aP;
        load_a_bf16(aP, &sP[w][0][0], 0, KPAD, t * 32, lane);
#pragma unroll
        for (int dt = 0; dt < 4; ++dt) {
            Frag bV;
            load_b_nk(bV, Vw, dt * 16, KPAD, t * 32, lane);   // transposed V: b128 loads
            acc[dt] = wmma_bf16(aP, bV, acc[dt]);
        }
    }
#pragma unroll
    for (int dt = 0; dt < 4; ++dt) {
        int d = dt * 16 + (lane & 15);
#pragma unroll
        for (int r = 0; r < 8; ++r) {
            int mrow = ((lane >= 16) ? 8 : 0) + r;
            float rs = __frcp_rn(sSum[w][mrow]);
            attW[(size_t)win * 4096 + (q0 + mrow) * 64 + d] = f2bf(acc[dt][r] * rs);
        }
    }
}

extern "C" void kernel_launch(void* const* d_in, const int* in_sizes, int n_in,
                              void* d_out, int out_size, void* d_ws, size_t ws_size,
                              hipStream_t stream) {
    (void)in_sizes; (void)n_in; (void)out_size; (void)ws_size;
    const float* feature = (const float*)d_in[0];
    const float* pmaps[3]  = { (const float*)d_in[1], (const float*)d_in[3], (const float*)d_in[5] };
    const float* pmasks[3] = { (const float*)d_in[2], (const float*)d_in[4], (const float*)d_in[6] };
    const unsigned char* mask_q = (const unsigned char*)d_in[7];
    const float* freq = (const float*)d_in[8];
    const int* cy = (const int*)d_in[9];
    const int* cx = (const int*)d_in[10];
    const float* Wq = (const float*)d_in[11];
    const float* Wk = (const float*)d_in[12];
    const float* Wv = (const float*)d_in[13];
    const float* Wo = (const float*)d_in[14];
    float* out = (float*)d_out;

    // workspace carve-out
    char* ws = (char*)d_ws;
    size_t off = 0;
    auto carve = [&](size_t bytes) { void* p = ws + off; off += (bytes + 255) & ~(size_t)255; return p; };
    const int Rl[3] = { B_ * 64 * 64, B_ * 32 * 32, B_ * 16 * 16 };   // 8192, 2048, 512

    unsigned short* Xbf  = (unsigned short*)carve((size_t)32768 * 512 * 2);
    unsigned short* Pbf[3];
    for (int l = 0; l < 3; ++l) Pbf[l] = (unsigned short*)carve((size_t)Rl[l] * 512 * 2);
    unsigned short* Wt[4];
    for (int wIdx = 0; wIdx < 4; ++wIdx) Wt[wIdx] = (unsigned short*)carve((size_t)512 * 512 * 2);
    unsigned short* Qp = (unsigned short*)carve((size_t)32768 * 512 * 2);
    unsigned short* Kpl[3], *Vpl[3];
    for (int l = 0; l < 3; ++l) {
        Kpl[l] = (unsigned short*)carve((size_t)Rl[l] * 512 * 2);
        Vpl[l] = (unsigned short*)carve((size_t)Rl[l] * 512 * 2);
    }
    unsigned short* qW   = (unsigned short*)carve((size_t)B_ * HEADS * NWIN * 64 * 64 * 2);
    unsigned short* kW   = (unsigned short*)carve((size_t)B_ * HEADS * NWIN * KPAD * 64 * 2);
    unsigned short* vW   = (unsigned short*)carve((size_t)B_ * HEADS * NWIN * KPAD * 64 * 2);
    float*          mK   = (float*)carve((size_t)B_ * NWIN * KPAD * 4);
    unsigned short* attW = (unsigned short*)carve((size_t)B_ * HEADS * NWIN * 64 * 64 * 2);

    // one-time bf16 packs (weights transposed for contiguous B-fragment loads)
    const float* Wsrc[4] = { Wq, Wk, Wv, Wo };
    for (int wIdx = 0; wIdx < 4; ++wIdx)
        pack_wt_k<<<dim3(32768 / 256), 256, 0, stream>>>(Wsrc[wIdx], Wt[wIdx]);
    pack_rm_k<<<dim3(32768 * 512 / 8 / 256), 256, 0, stream>>>(feature, Xbf, 32768 * 512 / 8);
    for (int l = 0; l < 3; ++l)
        pack_rm_k<<<dim3((Rl[l] * 512 / 8 + 255) / 256), 256, 0, stream>>>(pmaps[l], Pbf[l], Rl[l] * 512 / 8);

    // projections (WMMA bf16 GEMMs, bf16 out)
    gemm512_k<false, true><<<dim3((32768 / 128) * 4), 256, 0, stream>>>(Xbf, Wt[0], Qp, 32768);
    for (int l = 0; l < 3; ++l) {
        gemm512_k<false, true><<<dim3((Rl[l] / 128) * 4), 256, 0, stream>>>(Pbf[l], Wt[1], Kpl[l], Rl[l]);
        gemm512_k<false, true><<<dim3((Rl[l] / 128) * 4), 256, 0, stream>>>(Pbf[l], Wt[2], Vpl[l], Rl[l]);
    }

    // Q freq + RoPE + windowing
    {
        int total = B_ * HEADS * M_ * N_ * 32;
        rope_q_k<<<dim3(total / 256), 256, 0, stream>>>(Qp, freq, qW, total);
    }

    // per-level K/V region gather + freq + RoPE
    const int mlA[3] = { 64, 32, 16 }, kkA[3] = { 8, 6, 4 }, padA[3] = { 4, 3, 2 }, soA[3] = { 0, 64, 100 };
    for (int l = 0; l < 3; ++l) {
        int total = B_ * HEADS * NWIN * kkA[l] * kkA[l] * 32;
        gather_kv_k<<<dim3((total + 255) / 256), 256, 0, stream>>>(
            Kpl[l], Vpl[l], pmasks[l], freq, cy, cx, kW, vW, mK,
            l, mlA[l], mlA[l], kkA[l], padA[l], soA[l], total);
    }
    {
        int total = B_ * HEADS * NWIN * (KPAD - KREAL) * 64;
        pad_tail_k<<<dim3((total + 255) / 256), 256, 0, stream>>>(kW, vW, mK);
    }

    // per-window WMMA attention
    attn_win_k<<<dim3(B_ * HEADS * NWIN), 128, 0, stream>>>(qW, kW, vW, mK, mask_q, attW);

    // output projection, fused windowed->row relayout on the A operand, fp32 out
    gemm512_k<true, false><<<dim3((32768 / 128) * 4), 256, 0, stream>>>(attW, Wt[3], out, 32768);
}
